// GATEncoder_22892175687888
// MI455X (gfx1250) — matrix operand order repaired and checked by reading, not compiled, not run on hardware
//
#include <hip/hip_runtime.h>
#include <hip/hip_bf16.h>
#include <stdint.h>

// ---------------------------------------------------------------------------
// GAT forward for MI455X (gfx1250, wave32).  GEMMs via v_wmma_f32_16x16x32_bf16,
// edge softmax-aggregation via L2-resident float atomics (monotone-uint atomicMax
// for segment max, single fused pass for numerator+denominator).
// ---------------------------------------------------------------------------

typedef __bf16 bf16_t;
typedef __attribute__((ext_vector_type(16))) __bf16 v16bf;
typedef __attribute__((ext_vector_type(8)))  float  v8f;

#define NEG_SLOPE 0.2f
#define F_IN 128
#define HD1  256   // H*D of layer 1 (8*32)
#define H1   8
#define D1   32
#define F_OUT 64

// Monotone float <-> uint mapping so unsigned atomicMax == float max.
__device__ __forceinline__ unsigned fmap(float f) {
  unsigned u = __float_as_uint(f);
  return (u & 0x80000000u) ? ~u : (u | 0x80000000u);
}
__device__ __forceinline__ float funmap(unsigned u) {
  unsigned b = (u & 0x80000000u) ? (u & 0x7FFFFFFFu) : ~u;
  return __uint_as_float(b);
}

// ---------------------------------------------------------------------------
__global__ void k_zero4(float4* __restrict__ p, long long n4) {
  long long i = (long long)blockIdx.x * blockDim.x + threadIdx.x;
  if (i < n4) p[i] = make_float4(0.f, 0.f, 0.f, 0.f);
}

__global__ void k_f32_to_bf16(const float* __restrict__ s, bf16_t* __restrict__ d,
                              long long n) {
  long long i = (long long)blockIdx.x * blockDim.x + threadIdx.x;
  if (i < n) d[i] = (bf16_t)s[i];
}

// W: [K, Nc] f32 row-major  ->  Wt: [Nc, K] bf16 (so B-fragment K is contiguous)
__global__ void k_transpose_to_bf16(const float* __restrict__ W, bf16_t* __restrict__ Wt,
                                    int K, int Nc) {
  int i = blockIdx.x * blockDim.x + threadIdx.x;
  if (i >= K * Nc) return;
  int c = i / K, k = i - c * K;
  Wt[(size_t)c * K + k] = (bf16_t)W[(size_t)k * Nc + c];
}

// ---------------------------------------------------------------------------
// C[M,Nc] = A[M,K](bf16) * B[K,Nc]  with Bt = B^T [Nc,K](bf16).
// One wave -> one 16x16 tile, K-loop in steps of 32 via WMMA.
// A-frag layout (ISA 7.12.2, 16-bit A 16x32): lane<16: M=lane, K={0..7,16..23};
// lane>=16: M=lane-16, K={8..15,24..31}.
// B-frag: lane<16: N=lane, K={0..7,16..23}; lane>=16: N=lane-16, K={8..15,24..31}.
// C/D: VGPR r -> M = 16*mtile + 8*(lane>>4) + r, N = 16*ntile + (lane&15).
// ---------------------------------------------------------------------------
__global__ void k_gemm_wmma_bf16(const bf16_t* __restrict__ A,
                                 const bf16_t* __restrict__ Bt,
                                 float* __restrict__ C,
                                 int M, int K, int Nc) {
  const int lane = threadIdx.x & 31;
  const int wave = threadIdx.x >> 5;
  const int wavesPerBlk = blockDim.x >> 5;
  const int mtile = blockIdx.x;
  const int ntile = blockIdx.y * wavesPerBlk + wave;
  const int l15 = lane & 15;
  const int hi  = lane >> 4;

  int m = mtile * 16 + l15;
  if (m >= M) m = M - 1;                 // clamp (M is a multiple of 16 here anyway)
  const int n = ntile * 16 + l15;

  const bf16_t* ap = A  + (size_t)m * K + hi * 8;    // 16B aligned
  const bf16_t* bp = Bt + (size_t)n * K + hi * 8;    // 16B aligned

  v8f acc = {0.f, 0.f, 0.f, 0.f, 0.f, 0.f, 0.f, 0.f};

  for (int kk = 0; kk < K; kk += 32) {
    union { v16bf v; uint4 q[2]; } a, b;
    a.q[0] = *(const uint4*)(ap);        // K = kk + hi*8 + {0..7}
    a.q[1] = *(const uint4*)(ap + 16);   // K = kk + hi*8 + {16..23}
    b.q[0] = *(const uint4*)(bp);        // K = kk + hi*8 + {0..7}
    b.q[1] = *(const uint4*)(bp + 16);   // K = kk + hi*8 + {16..23}
    acc = __builtin_amdgcn_wmma_f32_16x16x32_bf16(
        /*neg_a=*/false, a.v, /*neg_b=*/false, b.v,
        /*c_mod=*/(short)0, acc, /*reuse_a=*/false, /*reuse_b=*/false);
    ap += 32; bp += 32;
  }

  const int mrow = mtile * 16 + hi * 8;
#pragma unroll
  for (int r = 0; r < 8; ++r) {
    int mm = mrow + r;
    if (mm < M) C[(size_t)mm * Nc + n] = acc[r];
  }
}

// ---------------------------------------------------------------------------
// Per-node attention dot products, layer 1: a_s[n,h] = <h1[n,h,:], att_src[h,:]>
__global__ void k_att1(const float* __restrict__ h1,
                       const float* __restrict__ asrc, const float* __restrict__ adst,
                       float* __restrict__ as1, float* __restrict__ ad1, int N) {
  int i = blockIdx.x * blockDim.x + threadIdx.x;   // i = n*8 + h
  if (i >= N * H1) return;
  int h = i & (H1 - 1);
  const float* hp = h1 + (size_t)i * D1;
  float s = 0.f, d = 0.f;
#pragma unroll 8
  for (int k = 0; k < D1; ++k) {
    float v = hp[k];
    s = fmaf(v, asrc[h * D1 + k], s);
    d = fmaf(v, adst[h * D1 + k], d);
  }
  as1[i] = s; ad1[i] = d;
}

// Layer 2 (H=1): a[n] = <h2[n,:], att[:]>
__global__ void k_att2(const float* __restrict__ h2,
                       const float* __restrict__ asrc, const float* __restrict__ adst,
                       float* __restrict__ as2, float* __restrict__ ad2, int N) {
  int n = blockIdx.x * blockDim.x + threadIdx.x;
  if (n >= N) return;
  const float* hp = h2 + (size_t)n * F_OUT;
  float s = 0.f, d = 0.f;
#pragma unroll 8
  for (int k = 0; k < F_OUT; ++k) {
    float v = hp[k];
    s = fmaf(v, asrc[k], s);
    d = fmaf(v, adst[k], d);
  }
  as2[n] = s; ad2[n] = d;
}

// ---------------------------------------------------------------------------
__device__ __forceinline__ void edge_sd(const int* __restrict__ ei, int e, int E,
                                        int& s, int& d) {
  if (e < E) { s = ei[e]; d = ei[E + e]; }
  else       { s = d = e - E; }          // self loop
}

// Segment max of leakyrelu(a_s[src]+a_d[dst]) per (dst, head), via uint atomicMax.
__global__ void k_edge_max1(const int* __restrict__ ei, int E, int N,
                            const float* __restrict__ as1, const float* __restrict__ ad1,
                            unsigned* __restrict__ emax) {
  long long t = (long long)blockIdx.x * blockDim.x + threadIdx.x;
  if (t >= (long long)(E + N) * H1) return;
  int e = (int)(t >> 3), h = (int)(t & 7);
  int s, d; edge_sd(ei, e, E, s, d);
  float v = as1[(size_t)s * H1 + h] + ad1[(size_t)d * H1 + h];
  v = v > 0.f ? v : NEG_SLOPE * v;
  atomicMax(emax + (size_t)d * H1 + h, fmap(v));
}

// Fused pass: denom[dst,h] += ex;  acc[dst,h,:] += ex * h1[src,h,:]
// 8 lanes per (edge, head); each lane does one float4 of the 32-wide feature.
__global__ void k_edge_msg1(const int* __restrict__ ei, int E, int N,
                            const float* __restrict__ as1, const float* __restrict__ ad1,
                            const unsigned* __restrict__ emax,
                            const float* __restrict__ h1,
                            float* __restrict__ den, float* __restrict__ acc) {
  long long t = (long long)blockIdx.x * blockDim.x + threadIdx.x;
  if (t >= (long long)(E + N) * 64) return;
  int e   = (int)(t >> 6);
  int sub = (int)(t & 63);
  int h   = sub >> 3;
  int dq  = (sub & 7) << 2;
  int s, d; edge_sd(ei, e, E, s, d);
  float v = as1[(size_t)s * H1 + h] + ad1[(size_t)d * H1 + h];
  v = v > 0.f ? v : NEG_SLOPE * v;
  float ex = __expf(v - funmap(emax[(size_t)d * H1 + h]));
  if ((sub & 7) == 0) atomicAdd(den + (size_t)d * H1 + h, ex);
  const float4 hv = *(const float4*)(h1 + ((size_t)s * H1 + h) * D1 + dq);
  float* ac = acc + ((size_t)d * H1 + h) * D1 + dq;
  atomicAdd(ac + 0, ex * hv.x);
  atomicAdd(ac + 1, ex * hv.y);
  atomicAdd(ac + 2, ex * hv.z);
  atomicAdd(ac + 3, ex * hv.w);
}

// out1 = elu(acc/denom + b1), emitted directly as bf16 (input of GEMM2)
__global__ void k_node1(const float* __restrict__ acc, const float* __restrict__ den,
                        const float* __restrict__ b1, bf16_t* __restrict__ o1b, int N) {
  long long i = (long long)blockIdx.x * blockDim.x + threadIdx.x;
  if (i >= (long long)N * HD1) return;
  int c = (int)(i & (HD1 - 1));
  int h = c >> 5;
  long long nn = i >> 8;
  float v = acc[i] / den[nn * H1 + h] + b1[c];
  v = v > 0.f ? v : (__expf(v) - 1.f);   // ELU(alpha=1)
  o1b[i] = (bf16_t)v;
}

__global__ void k_edge_max2(const int* __restrict__ ei, int E, int N,
                            const float* __restrict__ as2, const float* __restrict__ ad2,
                            unsigned* __restrict__ emax) {
  long long t = (long long)blockIdx.x * blockDim.x + threadIdx.x;
  if (t >= (long long)(E + N)) return;
  int e = (int)t;
  int s, d; edge_sd(ei, e, E, s, d);
  float v = as2[s] + ad2[d];
  v = v > 0.f ? v : NEG_SLOPE * v;
  atomicMax(emax + d, fmap(v));
}

// 16 lanes per edge, one float4 of the 64-wide feature each.
__global__ void k_edge_msg2(const int* __restrict__ ei, int E, int N,
                            const float* __restrict__ as2, const float* __restrict__ ad2,
                            const unsigned* __restrict__ emax,
                            const float* __restrict__ h2,
                            float* __restrict__ den, float* __restrict__ acc) {
  long long t = (long long)blockIdx.x * blockDim.x + threadIdx.x;
  if (t >= (long long)(E + N) * 16) return;
  int e   = (int)(t >> 4);
  int sub = (int)(t & 15);
  int dq  = sub << 2;
  int s, d; edge_sd(ei, e, E, s, d);
  float v = as2[s] + ad2[d];
  v = v > 0.f ? v : NEG_SLOPE * v;
  float ex = __expf(v - funmap(emax[d]));
  if (sub == 0) atomicAdd(den + d, ex);
  const float4 hv = *(const float4*)(h2 + (size_t)s * F_OUT + dq);
  float* ac = acc + (size_t)d * F_OUT + dq;
  atomicAdd(ac + 0, ex * hv.x);
  atomicAdd(ac + 1, ex * hv.y);
  atomicAdd(ac + 2, ex * hv.z);
  atomicAdd(ac + 3, ex * hv.w);
}

// Final: v = acc/den + b2, then log_softmax over 64 classes.
// One wave (32 lanes) per node, 2 classes per lane, shuffle reductions.
__global__ void k_final(const float* __restrict__ acc, const float* __restrict__ den,
                        const float* __restrict__ b2, float* __restrict__ out, int N) {
  int n = blockIdx.x * 8 + (threadIdx.x >> 5);
  int lane = threadIdx.x & 31;
  if (n >= N) return;
  float inv = 1.f / den[n];
  float v0 = acc[(size_t)n * F_OUT + lane]      * inv + b2[lane];
  float v1 = acc[(size_t)n * F_OUT + 32 + lane] * inv + b2[32 + lane];
  float mx = fmaxf(v0, v1);
#pragma unroll
  for (int o = 16; o; o >>= 1) mx = fmaxf(mx, __shfl_xor(mx, o, 32));
  float s = __expf(v0 - mx) + __expf(v1 - mx);
#pragma unroll
  for (int o = 16; o; o >>= 1) s += __shfl_xor(s, o, 32);
  float lse = mx + __logf(s);
  out[(size_t)n * F_OUT + lane]      = v0 - lse;
  out[(size_t)n * F_OUT + 32 + lane] = v1 - lse;
}

// ---------------------------------------------------------------------------
extern "C" void kernel_launch(void* const* d_in, const int* in_sizes, int n_in,
                              void* d_out, int out_size, void* d_ws, size_t ws_size,
                              hipStream_t stream) {
  const float* x        = (const float*)d_in[0];
  const int*   ei       = (const int*)  d_in[1];
  const float* W1       = (const float*)d_in[2];
  const float* att_src1 = (const float*)d_in[3];
  const float* att_dst1 = (const float*)d_in[4];
  const float* b1       = (const float*)d_in[5];
  const float* W2       = (const float*)d_in[6];
  const float* att_src2 = (const float*)d_in[7];
  const float* att_dst2 = (const float*)d_in[8];
  const float* b2       = (const float*)d_in[9];

  const int N = in_sizes[0] / F_IN;      // 50000
  const int E = in_sizes[1] / 2;         // 1600000
  const long long ET = (long long)E + N; // with self loops

  // ---- workspace layout (bump allocator, 256B aligned) ----
  char* ws = (char*)d_ws;
  size_t off = 0;
  auto alloc = [&](size_t bytes) -> char* {
    char* p = ws + off;
    off = (off + bytes + 255) & ~(size_t)255;
    return p;
  };
  bf16_t*   xb   = (bf16_t*)  alloc((size_t)N * F_IN * 2);
  bf16_t*   w1t  = (bf16_t*)  alloc((size_t)HD1 * F_IN * 2);   // W1^T [256,128]
  bf16_t*   w2t  = (bf16_t*)  alloc((size_t)F_OUT * HD1 * 2);  // W2^T [64,256]
  float*    h1   = (float*)   alloc((size_t)N * HD1 * 4);
  float*    as1  = (float*)   alloc((size_t)N * H1 * 4);
  float*    ad1  = (float*)   alloc((size_t)N * H1 * 4);
  // zero-region 1 (contiguous): emax1 | den1 | acc1
  unsigned* emax1 = (unsigned*)alloc((size_t)N * H1 * 4);
  float*    den1  = (float*)   alloc((size_t)N * H1 * 4);
  float*    acc1  = (float*)   alloc((size_t)N * HD1 * 4);
  bf16_t*   o1b   = (bf16_t*)  alloc((size_t)N * HD1 * 2);
  float*    h2    = (float*)   alloc((size_t)N * F_OUT * 4);
  float*    as2   = (float*)   alloc((size_t)N * 4);
  float*    ad2   = (float*)   alloc((size_t)N * 4);
  // zero-region 2 (contiguous): emax2 | den2 | acc2
  unsigned* emax2 = (unsigned*)alloc((size_t)N * 4);
  float*    den2  = (float*)   alloc((size_t)N * 4);
  float*    acc2  = (float*)   alloc((size_t)N * F_OUT * 4);
  (void)ws_size;

  const int TB = 256;
  auto blocks = [&](long long n, int tpb) -> unsigned {
    return (unsigned)((n + tpb - 1) / tpb);
  };

  // 0. zero accumulators (fmap(-inf) > 0, so 0 is the atomicMax identity)
  long long z1f = (long long)N * H1 * 2 + (long long)N * HD1;   // floats in region 1
  long long z2f = (long long)N * 2 + (long long)N * F_OUT;      // floats in region 2
  k_zero4<<<blocks(z1f / 4, TB), TB, 0, stream>>>((float4*)emax1, z1f / 4);
  k_zero4<<<blocks(z2f / 4, TB), TB, 0, stream>>>((float4*)emax2, z2f / 4);

  // 1. precision conversions (feed the WMMA path)
  k_f32_to_bf16<<<blocks((long long)N * F_IN, TB), TB, 0, stream>>>(x, xb, (long long)N * F_IN);
  k_transpose_to_bf16<<<blocks((long long)F_IN * HD1, TB), TB, 0, stream>>>(W1, w1t, F_IN, HD1);
  k_transpose_to_bf16<<<blocks((long long)HD1 * F_OUT, TB), TB, 0, stream>>>(W2, w2t, HD1, F_OUT);

  // 2. layer 1 GEMM: h1[N,256] = x @ W1   (8 waves/block -> 8 N-tiles, grid.y=2)
  const int mtiles = (N + 15) / 16;
  k_gemm_wmma_bf16<<<dim3(mtiles, 2), 256, 0, stream>>>(xb, w1t, h1, N, F_IN, HD1);

  // 3. attention logits per node
  k_att1<<<blocks((long long)N * H1, TB), TB, 0, stream>>>(h1, att_src1, att_dst1, as1, ad1, N);

  // 4. segment max, then fused numerator/denominator accumulation (L2 resident)
  k_edge_max1<<<blocks(ET * H1, TB), TB, 0, stream>>>(ei, E, N, as1, ad1, emax1);
  k_edge_msg1<<<blocks(ET * 64, TB), TB, 0, stream>>>(ei, E, N, as1, ad1, emax1, h1, den1, acc1);

  // 5. normalize + bias + ELU -> bf16 activations for layer 2
  k_node1<<<blocks((long long)N * HD1, TB), TB, 0, stream>>>(acc1, den1, b1, o1b, N);

  // 6. layer 2 GEMM: h2[N,64] = o1 @ W2   (4 waves/block -> 4 N-tiles)
  k_gemm_wmma_bf16<<<dim3(mtiles, 1), 128, 0, stream>>>(o1b, w2t, h2, N, HD1, F_OUT);

  // 7. layer-2 attention + aggregation
  k_att2<<<blocks(N, TB), TB, 0, stream>>>(h2, att_src2, att_dst2, as2, ad2, N);
  k_edge_max2<<<blocks(ET, TB), TB, 0, stream>>>(ei, E, N, as2, ad2, emax2);
  k_edge_msg2<<<blocks(ET * 16, TB), TB, 0, stream>>>(ei, E, N, as2, ad2, emax2, h2, den2, acc2);

  // 8. bias + log_softmax (one wave32 per node)
  k_final<<<blocks((N + 7) / 8, 1), 256, 0, stream>>>(acc2, den2, b2, (float*)d_out, N);
}
